// MANA_25262997635097
// MI455X (gfx1250) — matrix operand order repaired and checked by reading, not compiled, not run on hardware
//
#include <hip/hip_runtime.h>
#include <hip/hip_bf16.h>
#include <math.h>

// ---------------------------------------------------------------------------
// PaiNN-style GNN forward for MI455X (gfx1250, wave32, WMMA).
//
// Input index map (setup_inputs() insertion order; 'params' flattened as
// pytree leaves, dict keys alphabetical):
//   0:z(N i32) 1:edge_index(2E i32) 2:edge_attr(E*4 f32) 3:batch(N i32)
//   4:dielectric(B f32)
//   5:centers(20) 6:embed(100*64) 7:gamma(20)
//   8:lam.W1(64*64) 9:lam.W2(64) 10:lam.b1(64) 11:lam.b2(1)
//   12+8l..19+8l (l=0..3): fW1(20*64) fW2(64*192) fb1(64) fb2(192)
//                           uW1(192*64) uW2(64*192) ub1(64) ub2(192)
//   44:phi.W1(128*128) 45:phi.W2(128*64) 46:phi.W3(64) 47:phi.b1(128)
//   48:phi.b2(64) 49:phi.b3(1) 50:phi.ln_b(128) 51:phi.ln_w(128)
//   52:solv.W1(64) 53:solv.W2(64*64) 54:solv.b1(64) 55:solv.b2(64)
// Output: d_out[0..255]=lam, d_out[256..511]=softplus head.
// ---------------------------------------------------------------------------

typedef __attribute__((ext_vector_type(16))) _Float16 v16h;
typedef __attribute__((ext_vector_type(8)))  _Float16 v8h;
typedef __attribute__((ext_vector_type(8)))  float    v8f;

#define NNODE  10000
#define NEDGE  320000
#define FDIM   64
#define RDIM   20
#define NLAYER 4
#define NB     256

// silu via v_exp_f32 + v_rcp_f32 (avoids the IEEE v_div_scale/div_fmas chain)
__device__ __forceinline__ float silu_f(float x) {
  return x * __builtin_amdgcn_rcpf(1.0f + __expf(-x));
}

union AFrag { v16h v; v8h h[2]; };

// ---------------- precompute kernels ----------------

// rbf: E x 32 f16, padded K=20..31 with zeros; row = 64B (two 16B chunks / lane half)
__global__ __launch_bounds__(256) void k_rbf(const float* __restrict__ ea,
                                             const float* __restrict__ centers,
                                             const float* __restrict__ gamma,
                                             _Float16* __restrict__ rbf) {
  int e = blockIdx.x * blockDim.x + threadIdx.x;
  if (e >= NEDGE) return;
  float d = ea[e * 4 + 0];
  _Float16* out = rbf + (size_t)e * 32;
  #pragma unroll
  for (int k = 0; k < 32; ++k) {
    float val = 0.0f;
    if (k < RDIM) {
      float t = d - centers[k];
      val = __expf(-gamma[k] * t * t);
    }
    out[k] = (_Float16)val;
  }
}

__global__ __launch_bounds__(256) void k_init_s(const int* __restrict__ z,
                                                const float* __restrict__ embed,
                                                float* __restrict__ s) {
  int t = blockIdx.x * blockDim.x + threadIdx.x;
  if (t >= NNODE * 64) return;
  int n = t >> 6, f = t & 63;
  s[t] = embed[z[n] * 64 + f];
}

// Pack a K x Ncols f32 row-major weight into WMMA B-fragments (f16):
// fragment (kc,nt), lane l: holds W[kc*32 + (j<8 ? j : j+8) + 8*(l/16)][nt*16 + (l%16)], j=0..15
__global__ void k_pack(const float* __restrict__ W, int Krows, int KC, int NT,
                       int Ncols, _Float16* __restrict__ dst) {
  int t = blockIdx.x * blockDim.x + threadIdx.x;
  if (t >= KC * NT * 32) return;
  int lane = t & 31, frag = t >> 5;
  int nt = frag % NT, kc = frag / NT;
  int l0 = lane & 15, hi = lane >> 4;
  int n = nt * 16 + l0;
  _Float16* o = dst + ((size_t)(frag * 32 + lane) << 4);
  #pragma unroll
  for (int j = 0; j < 16; ++j) {
    int k = kc * 32 + (j < 8 ? j : j + 8) + 8 * hi;
    float val = (k < Krows) ? W[k * Ncols + n] : 0.0f;
    o[j] = (_Float16)val;
  }
}

// ---------------- per-layer fused edge kernel ----------------
// Block = 256 thr = 8 waves; each wave owns 16 edges. GEMM1 (K=32,N=64) ->
// silu -> LDS re-layout -> GEMM2 (K=64,N=192) -> message+atomic scatter
// straight from the 12 accumulator tiles. grid = E/128 = 2500 (exact).
__global__ __launch_bounds__(256) void k_edge(
    const _Float16* __restrict__ rbf, const _Float16* __restrict__ wf1,
    const _Float16* __restrict__ wf2, const float* __restrict__ fb1,
    const float* __restrict__ fb2, const int* __restrict__ rowIdx,
    const int* __restrict__ colIdx, const float* __restrict__ ea,
    const float* __restrict__ s, const float* __restrict__ v,
    float* __restrict__ m_s, float* __restrict__ m_v) {
  __shared__ __align__(16) _Float16 ldsH[8][16][64];
  int tid = threadIdx.x;
  int wave = tid >> 5, lane = tid & 31;
  int l0 = lane & 15, hi = lane >> 4;
  int eBase = blockIdx.x * 128 + wave * 16;

  // A1 fragment: 16 rbf rows. A-layout: half j -> K = (j<8 ? j : j+8) + 8*hi
  AFrag a1;
  {
    const _Float16* p = rbf + (size_t)(eBase + l0) * 32;
    a1.h[0] = *(const v8h*)(p + 8 * hi);
    a1.h[1] = *(const v8h*)(p + 16 + 8 * hi);
  }
  // GEMM1: phi1 = silu(rbf @ fW1 + fb1), 16x64
  #pragma unroll
  for (int nt = 0; nt < 4; ++nt) {
    v16h b = *(const v16h*)(wf1 + ((size_t)(nt * 32 + lane) << 4));
    v8f c = {};
    c = __builtin_amdgcn_wmma_f32_16x16x32_f16(false, a1.v, false, b, (short)0, c,
                                               false, false);
    int n = nt * 16 + l0;
    float bias = fb1[n];
    #pragma unroll
    for (int r = 0; r < 8; ++r)
      ldsH[wave][r + 8 * hi][n] = (_Float16)silu_f(c[r] + bias);
  }
  // Re-load as A fragments (K=64 -> 2 chunks). DS ops are in-order per wave.
  AFrag a2[2];
  #pragma unroll
  for (int kc = 0; kc < 2; ++kc) {
    const _Float16* p = &ldsH[wave][l0][kc * 32 + 8 * hi];
    a2[kc].h[0] = *(const v8h*)p;
    a2[kc].h[1] = *(const v8h*)(p + 16);
  }
  // GEMM2: phi = phi1 @ fW2, 16x192 (12 N-tiles x 2 K-chunks)
  v8f acc[12];
  #pragma unroll
  for (int nt = 0; nt < 12; ++nt) {
    v8f c = {};
    c = __builtin_amdgcn_wmma_f32_16x16x32_f16(
        false, a2[0].v, false,
        *(const v16h*)(wf2 + ((size_t)((0 * 12 + nt) * 32 + lane) << 4)), (short)0,
        c, false, false);
    c = __builtin_amdgcn_wmma_f32_16x16x32_f16(
        false, a2[1].v, false,
        *(const v16h*)(wf2 + ((size_t)((1 * 12 + nt) * 32 + lane) << 4)), (short)0,
        c, false, false);
    acc[nt] = c;
  }
  // Message + scatter. C layout: lane col n = nt*16+l0 ; VGPR r -> row r+8*hi.
  float b_ss[4], b_vv[4], b_sv[4];
  #pragma unroll
  for (int j = 0; j < 4; ++j) {
    int f = j * 16 + l0;
    b_ss[j] = fb2[f]; b_vv[j] = fb2[64 + f]; b_sv[j] = fb2[128 + f];
  }
  #pragma unroll
  for (int r = 0; r < 8; ++r) {
    int e = eBase + r + 8 * hi;
    int dst = rowIdx[e], src = colIdx[e];
    float dx = ea[e * 4 + 1], dy = ea[e * 4 + 2], dz = ea[e * 4 + 3];
    #pragma unroll
    for (int j = 0; j < 4; ++j) {
      int f = j * 16 + l0;
      float ss = acc[j][r] + b_ss[j];
      float vv = acc[4 + j][r] + b_vv[j];
      float sv = acc[8 + j][r] + b_sv[j];
      float sc = s[src * 64 + f];
      unsafeAtomicAdd(&m_s[dst * 64 + f], ss * sc);
      const float* vp = v + (size_t)src * 192 + f * 3;
      float* mp = m_v + (size_t)dst * 192 + f * 3;
      unsafeAtomicAdd(mp + 0, vv * vp[0] + sv * sc * dx);
      unsafeAtomicAdd(mp + 1, vv * vp[1] + sv * sc * dy);
      unsafeAtomicAdd(mp + 2, vv * vp[2] + sv * sc * dz);
    }
  }
}

// ---------------- per-layer node update kernel ----------------
// Wave owns 16 nodes: x=[s,m_s,|m_v|] (192) -> 64 -> 192; apply s += ds,
// v = alpha*v + beta*m_v directly from fragments (no atomics needed).
__global__ __launch_bounds__(256) void k_update(
    const _Float16* __restrict__ wu1, const _Float16* __restrict__ wu2,
    const float* __restrict__ ub1, const float* __restrict__ ub2,
    float* __restrict__ s, float* __restrict__ v, const float* __restrict__ m_s,
    const float* __restrict__ m_v) {
  __shared__ __align__(16) _Float16 ldsX[8][16][192];
  __shared__ __align__(16) _Float16 ldsH[8][16][64];
  int tid = threadIdx.x;
  int wave = tid >> 5, lane = tid & 31;
  int l0 = lane & 15, hi = lane >> 4;
  int nBase = blockIdx.x * 128 + wave * 16;
  if (nBase >= NNODE) return;  // N is a multiple of 16; no partial waves

  for (int i = lane; i < 16 * 192; i += 32) {
    int node = nBase + (i / 192), f = i % 192;
    float val;
    if (f < 64) val = s[node * 64 + f];
    else if (f < 128) val = m_s[node * 64 + (f - 64)];
    else {
      const float* mp = m_v + (size_t)node * 192 + (f - 128) * 3;
      val = sqrtf(mp[0] * mp[0] + mp[1] * mp[1] + mp[2] * mp[2] + 1e-12f);
    }
    ldsX[wave][i / 192][f] = (_Float16)val;
  }
  AFrag ax[6];
  #pragma unroll
  for (int kc = 0; kc < 6; ++kc) {
    const _Float16* p = &ldsX[wave][l0][kc * 32 + 8 * hi];
    ax[kc].h[0] = *(const v8h*)p;
    ax[kc].h[1] = *(const v8h*)(p + 16);
  }
  #pragma unroll
  for (int nt = 0; nt < 4; ++nt) {
    v8f c = {};
    #pragma unroll
    for (int kc = 0; kc < 6; ++kc)
      c = __builtin_amdgcn_wmma_f32_16x16x32_f16(
          false, ax[kc].v, false,
          *(const v16h*)(wu1 + ((size_t)((kc * 4 + nt) * 32 + lane) << 4)),
          (short)0, c, false, false);
    int n = nt * 16 + l0;
    float bias = ub1[n];
    #pragma unroll
    for (int r = 0; r < 8; ++r)
      ldsH[wave][r + 8 * hi][n] = (_Float16)silu_f(c[r] + bias);
  }
  AFrag a2[2];
  #pragma unroll
  for (int kc = 0; kc < 2; ++kc) {
    const _Float16* p = &ldsH[wave][l0][kc * 32 + 8 * hi];
    a2[kc].h[0] = *(const v8h*)p;
    a2[kc].h[1] = *(const v8h*)(p + 16);
  }
  v8f u[12];
  #pragma unroll
  for (int nt = 0; nt < 12; ++nt) {
    v8f c = {};
    c = __builtin_amdgcn_wmma_f32_16x16x32_f16(
        false, a2[0].v, false,
        *(const v16h*)(wu2 + ((size_t)((0 * 12 + nt) * 32 + lane) << 4)), (short)0,
        c, false, false);
    c = __builtin_amdgcn_wmma_f32_16x16x32_f16(
        false, a2[1].v, false,
        *(const v16h*)(wu2 + ((size_t)((1 * 12 + nt) * 32 + lane) << 4)), (short)0,
        c, false, false);
    u[nt] = c;
  }
  #pragma unroll
  for (int r = 0; r < 8; ++r) {
    int node = nBase + r + 8 * hi;
    #pragma unroll
    for (int j = 0; j < 4; ++j) {
      int f = j * 16 + l0;
      float ds = u[j][r] + ub2[f];
      float al = u[4 + j][r] + ub2[64 + f];
      float be = u[8 + j][r] + ub2[128 + f];
      s[node * 64 + f] += ds;
      size_t base = (size_t)node * 192 + f * 3;
      #pragma unroll
      for (int c3 = 0; c3 < 3; ++c3)
        v[base + c3] = al * v[base + c3] + be * m_v[base + c3];
    }
  }
}

// ---------------- readout ----------------

__global__ __launch_bounds__(256) void k_accum(const float* __restrict__ s,
                                               const int* __restrict__ batch,
                                               float* __restrict__ h_mol,
                                               float* __restrict__ counts) {
  int t = blockIdx.x * blockDim.x + threadIdx.x;
  if (t >= NNODE * 64) return;
  int n = t >> 6, f = t & 63;
  int b = batch[n];
  unsafeAtomicAdd(&h_mol[b * 64 + f], s[t]);
  if (f == 0) unsafeAtomicAdd(&counts[b], 1.0f);
}

__global__ __launch_bounds__(256) void k_readout(
    const float* __restrict__ h_mol, const float* __restrict__ counts,
    const float* __restrict__ diel, const float* __restrict__ lamW1,
    const float* __restrict__ lamW2, const float* __restrict__ lamb1,
    const float* __restrict__ lamb2, const float* __restrict__ soW1,
    const float* __restrict__ soW2, const float* __restrict__ sob1,
    const float* __restrict__ sob2, const float* __restrict__ pW1,
    const float* __restrict__ pW2, const float* __restrict__ pW3,
    const float* __restrict__ pb1, const float* __restrict__ pb2,
    const float* __restrict__ pb3, const float* __restrict__ ln_w,
    const float* __restrict__ ln_b, float* __restrict__ out) {
  int b = blockIdx.x * blockDim.x + threadIdx.x;
  if (b >= NB) return;
  float h[128], t1[64];
  float inv = __builtin_amdgcn_rcpf(counts[b] + 1e-9f);
  for (int f = 0; f < 64; ++f) h[f] = h_mol[b * 64 + f] * inv;
  // lam head
  for (int n = 0; n < 64; ++n) {
    float a = lamb1[n];
    for (int k = 0; k < 64; ++k) a += h[k] * lamW1[k * 64 + n];
    t1[n] = silu_f(a);
  }
  float lam = lamb2[0];
  for (int n = 0; n < 64; ++n) lam += t1[n] * lamW2[n];
  // solvent MLP
  float d = diel[b];
  for (int n = 0; n < 64; ++n) t1[n] = silu_f(d * soW1[n] + sob1[n]);
  for (int n = 0; n < 64; ++n) {
    float a = sob2[n];
    for (int k = 0; k < 64; ++k) a += t1[k] * soW2[k * 64 + n];
    h[64 + n] = a;
  }
  // phi head
  float x[128];
  for (int n = 0; n < 128; ++n) {
    float a = pb1[n];
    for (int k = 0; k < 128; ++k) a += h[k] * pW1[k * 128 + n];
    x[n] = silu_f(a);
  }
  float mu = 0.0f;
  for (int n = 0; n < 128; ++n) mu += x[n];
  mu *= (1.0f / 128.0f);
  float var = 0.0f;
  for (int n = 0; n < 128; ++n) { float dd = x[n] - mu; var += dd * dd; }
  var *= (1.0f / 128.0f);
  float rstd = rsqrtf(var + 1e-5f);
  for (int n = 0; n < 128; ++n) x[n] = (x[n] - mu) * rstd * ln_w[n] + ln_b[n];
  float y;
  float o = pb3[0];
  for (int n = 0; n < 64; ++n) {
    float a = pb2[n];
    for (int k = 0; k < 128; ++k) a += x[k] * pW2[k * 64 + n];
    y = silu_f(a);
    o += y * pW3[n];
  }
  float sp = (o > 20.0f) ? o : log1pf(__expf(o));
  out[b] = lam;
  out[256 + b] = sp;
}

// ---------------- host launch ----------------

extern "C" void kernel_launch(void* const* d_in, const int* in_sizes, int n_in,
                              void* d_out, int out_size, void* d_ws, size_t ws_size,
                              hipStream_t stream) {
  const int* z     = (const int*)d_in[0];
  const int* ei    = (const int*)d_in[1];  // [2,E]
  const float* ea  = (const float*)d_in[2];
  const int* batch = (const int*)d_in[3];
  const float* diel = (const float*)d_in[4];
  const float* centers = (const float*)d_in[5];
  const float* embed   = (const float*)d_in[6];
  const float* gamma   = (const float*)d_in[7];
  const float* lamW1 = (const float*)d_in[8];
  const float* lamW2 = (const float*)d_in[9];
  const float* lamb1 = (const float*)d_in[10];
  const float* lamb2 = (const float*)d_in[11];
  const float* pW1 = (const float*)d_in[44];
  const float* pW2 = (const float*)d_in[45];
  const float* pW3 = (const float*)d_in[46];
  const float* pb1 = (const float*)d_in[47];
  const float* pb2 = (const float*)d_in[48];
  const float* pb3 = (const float*)d_in[49];
  const float* ln_b = (const float*)d_in[50];
  const float* ln_w = (const float*)d_in[51];
  const float* soW1 = (const float*)d_in[52];
  const float* soW2 = (const float*)d_in[53];
  const float* sob1 = (const float*)d_in[54];
  const float* sob2 = (const float*)d_in[55];

  // workspace layout (bytes)
  char* ws = (char*)d_ws;
  size_t off = 0;
  _Float16* rbf = (_Float16*)(ws + off); off += (size_t)NEDGE * 32 * 2;   // 20.48 MB
  float* s   = (float*)(ws + off); off += (size_t)NNODE * 64 * 4;          // 2.56 MB
  float* v   = (float*)(ws + off); off += (size_t)NNODE * 192 * 4;         // 7.68 MB
  float* m_s = (float*)(ws + off); off += (size_t)NNODE * 64 * 4;          // 2.56 MB
  float* m_v = (float*)(ws + off); off += (size_t)NNODE * 192 * 4;         // 7.68 MB
  char* wfrag = ws + off; off += (size_t)NLAYER * 77824;                   // 0.31 MB
  float* h_mol  = (float*)(ws + off); off += (size_t)NB * 64 * 4;
  float* counts = (float*)(ws + off); off += (size_t)NB * 4;
  // total ~41.4 MB, assumed <= ws_size

  // precompute
  k_rbf<<<(NEDGE + 255) / 256, 256, 0, stream>>>(ea, centers, gamma, rbf);
  k_init_s<<<(NNODE * 64 + 255) / 256, 256, 0, stream>>>(z, embed, s);
  hipMemsetAsync(v, 0, (size_t)NNODE * 192 * 4, stream);

  for (int l = 0; l < NLAYER; ++l) {
    const float* fW1 = (const float*)d_in[12 + 8 * l];
    const float* fW2 = (const float*)d_in[13 + 8 * l];
    const float* uW1 = (const float*)d_in[16 + 8 * l];
    const float* uW2 = (const float*)d_in[17 + 8 * l];
    char* base = wfrag + (size_t)l * 77824;
    k_pack<<<1, 128, 0, stream>>>(fW1, RDIM, 1, 4, 64, (_Float16*)(base + 0));
    k_pack<<<3, 256, 0, stream>>>(fW2, 64, 2, 12, 192, (_Float16*)(base + 4096));
    k_pack<<<3, 256, 0, stream>>>(uW1, 192, 6, 4, 64, (_Float16*)(base + 28672));
    k_pack<<<3, 256, 0, stream>>>(uW2, 64, 2, 12, 192, (_Float16*)(base + 53248));
  }

  for (int l = 0; l < NLAYER; ++l) {
    const float* fb1 = (const float*)d_in[14 + 8 * l];
    const float* fb2 = (const float*)d_in[15 + 8 * l];
    const float* ub1 = (const float*)d_in[18 + 8 * l];
    const float* ub2 = (const float*)d_in[19 + 8 * l];
    char* base = wfrag + (size_t)l * 77824;
    hipMemsetAsync(m_s, 0, (size_t)NNODE * 64 * 4, stream);
    hipMemsetAsync(m_v, 0, (size_t)NNODE * 192 * 4, stream);
    k_edge<<<NEDGE / 128, 256, 0, stream>>>(
        rbf, (const _Float16*)(base + 0), (const _Float16*)(base + 4096), fb1, fb2,
        ei, ei + NEDGE, ea, s, v, m_s, m_v);
    k_update<<<(NNODE + 127) / 128, 256, 0, stream>>>(
        (const _Float16*)(base + 28672), (const _Float16*)(base + 53248), ub1, ub2,
        s, v, m_s, m_v);
  }

  hipMemsetAsync(h_mol, 0, (size_t)NB * 64 * 4, stream);
  hipMemsetAsync(counts, 0, (size_t)NB * 4, stream);
  k_accum<<<(NNODE * 64 + 255) / 256, 256, 0, stream>>>(s, batch, h_mol, counts);
  k_readout<<<1, 256, 0, stream>>>(h_mol, counts, diel, lamW1, lamW2, lamb1, lamb2,
                                   soW1, soW2, sob1, sob2, pW1, pW2, pW3, pb1, pb2,
                                   pb3, ln_w, ln_b, (float*)d_out);
}
